// GCN_15685220565562
// MI455X (gfx1250) — compile-verified, matrix operand order checked by reading, not run on hardware
//
#include <hip/hip_runtime.h>
#include <hip/hip_fp16.h>

typedef __attribute__((ext_vector_type(16))) _Float16 v16h;
typedef __attribute__((ext_vector_type(8)))  _Float16 v8h;
typedef __attribute__((ext_vector_type(8)))  float    v8f;

#define DEVFN static __device__ __forceinline__

constexpr int NN  = 50000;   // nodes
constexpr int NE  = 300000;  // edges
constexpr int NF  = 128;     // node feature dim
constexpr int EF  = 16;      // edge feature dim
constexpr int HID = 256;
constexpr int Mp  = 50048;   // NN padded to 128
constexpr int Ep  = 300032;  // NE padded to 128
constexpr int KC  = 544;     // concat K (2*HID+EF=528) padded to mult of 32

// ---------------------------------------------------------------- prep kernels

__global__ void k_fill_deg(float* deg) {
    int i = blockIdx.x * 256 + threadIdx.x;
    if (i < NN) deg[i] = 1.0f;                         // self-loop
}

__global__ void k_deg_scatter(const int* __restrict__ dst, float* deg) {
    int e = blockIdx.x * 256 + threadIdx.x;
    if (e < NE) unsafeAtomicAdd(&deg[dst[e]], 1.0f);   // global_atomic_add_f32
}

__global__ void k_node_norm(const float* __restrict__ deg, float* dinv, float* idg) {
    int i = blockIdx.x * 256 + threadIdx.x;
    if (i < NN) { float d = deg[i]; dinv[i] = rsqrtf(d); idg[i] = 1.0f / d; }
}

__global__ void k_edge_norm(const int* __restrict__ src, const int* __restrict__ dst,
                            const float* __restrict__ dinv, float* nrme) {
    int e = blockIdx.x * 256 + threadIdx.x;
    if (e < NE) nrme[e] = dinv[src[e]] * dinv[dst[e]];
}

// W[K,N] f32 -> Wt[N,Kp] f16 (transposed, k-padded with zeros)
__global__ void k_wt(const float* __restrict__ W, _Float16* __restrict__ Wt,
                     int K, int N, int Kp) {
    int i = blockIdx.x * 256 + threadIdx.x;
    if (i >= N * Kp) return;
    int n = i / Kp, k = i % Kp;
    Wt[i] = (k < K) ? (_Float16)W[(size_t)k * N + n] : (_Float16)0.0f;
}

// X[M,C] f32 -> Y[Mp,C] f16, pad rows zeroed
__global__ void k_cvt_rows(const float* __restrict__ X, _Float16* __restrict__ Y,
                           int M, int C, long total) {
    long i = (long)blockIdx.x * 256 + threadIdx.x;
    if (i >= total) return;
    long row = i / C;
    Y[i] = (row < M) ? (_Float16)X[i] : (_Float16)0.0f;
}

// ---------------------------------------------------------------- fragment helpers

DEVFN v16h loadfrag(const _Float16* p) {   // two contiguous b128 loads per lane
    v8h lo = *(const v8h*)p;
    v8h hi = *(const v8h*)(p + 16);
    return __builtin_shufflevector(lo, hi, 0,1,2,3,4,5,6,7,8,9,10,11,12,13,14,15);
}

// ---------------------------------------------------------------- WMMA GEMM
// C[M,N] = act(A[M,K] @ W[K,N] + b). A f16 row-major, Wt = W^T f16 [N][K].
// Block = 256 thr (8 waves); wave -> 16 rows x 64 cols (4 accumulators).
// Software-pipelined: k+1 fragments are fetched before k's WMMAs consume k's.
template<int K, bool RELU>
__global__ __launch_bounds__(256) void gemm_f16(
    const _Float16* __restrict__ A, const _Float16* __restrict__ Wt,
    const float* __restrict__ bias, _Float16* __restrict__ out, int N) {
    const int lane  = threadIdx.x & 31;
    const int wave  = threadIdx.x >> 5;
    const int mrow  = blockIdx.x * 128 + wave * 16;
    const int ntile = blockIdx.y * 64;
    const int lm    = lane & 15;
    const int kg    = (lane >> 4) * 8;          // k-group per ISA 16-bit A layout
    const _Float16* arow = A  + (size_t)(mrow + lm) * K + kg;
    const _Float16* brow = Wt + (size_t)(ntile + lm) * K + kg;   // +16*K per n-tile

    v8f acc[4];
#pragma unroll
    for (int t = 0; t < 4; ++t) acc[t] = {};

    // prologue: fetch k-step 0
    v16h a  = loadfrag(arow);
    v16h b0 = loadfrag(brow);
    v16h b1 = loadfrag(brow + (size_t)16 * K);
    v16h b2 = loadfrag(brow + (size_t)32 * K);
    v16h b3 = loadfrag(brow + (size_t)48 * K);

#pragma unroll
    for (int kk = 32; kk <= K; kk += 32) {
        v16h an = a, bn0 = b0, bn1 = b1, bn2 = b2, bn3 = b3;
        if (kk < K) {                          // prefetch next k-step
            an  = loadfrag(arow + kk);
            bn0 = loadfrag(brow + kk);
            bn1 = loadfrag(brow + (size_t)16 * K + kk);
            bn2 = loadfrag(brow + (size_t)32 * K + kk);
            bn3 = loadfrag(brow + (size_t)48 * K + kk);
        }
        acc[0] = __builtin_amdgcn_wmma_f32_16x16x32_f16(false, a, false, b0, (short)0, acc[0], false, false);
        acc[1] = __builtin_amdgcn_wmma_f32_16x16x32_f16(false, a, false, b1, (short)0, acc[1], false, false);
        acc[2] = __builtin_amdgcn_wmma_f32_16x16x32_f16(false, a, false, b2, (short)0, acc[2], false, false);
        acc[3] = __builtin_amdgcn_wmma_f32_16x16x32_f16(false, a, false, b3, (short)0, acc[3], false, false);
        a = an; b0 = bn0; b1 = bn1; b2 = bn2; b3 = bn3;
    }

    // C/D layout: VGPR v -> row v + 8*(lane>=16), col = lane&15
    const int orow = mrow + (lane >> 4) * 8;
#pragma unroll
    for (int t = 0; t < 4; ++t) {
        int col = ntile + 16 * t + lm;
        float bv = bias[col];
#pragma unroll
        for (int v = 0; v < 8; ++v) {
            float x = acc[t][v] + bv;
            if (RELU) x = fmaxf(x, 0.0f);
            out[(size_t)(orow + v) * N + col] = (_Float16)x;
        }
    }
}

// 8-half group loader for concat row [x_src(256) | x_dst(256) | edge_attr(16) | 0-pad]
DEVFN v8h load8_concat(const _Float16* xs, const _Float16* xd, const float* ea, int k) {
    if (k < 256) return *(const v8h*)(xs + k);
    if (k < 512) return *(const v8h*)(xd + (k - 256));
    v8h r = {};
    if (k < 528) {
#pragma unroll
        for (int i = 0; i < 8; ++i) r[i] = (_Float16)ea[(k - 512) + i];
    }
    return r;
}

DEVFN v16h loadA_concat(const _Float16* xs, const _Float16* xd, const float* ea, int k) {
    v8h lo = load8_concat(xs, xd, ea, k);
    v8h hi = load8_concat(xs, xd, ea, k + 16);
    return __builtin_shufflevector(lo, hi, 0,1,2,3,4,5,6,7,8,9,10,11,12,13,14,15);
}

// classifier layer 0: H1[e] = relu(concat(x[src],x[dst],ea) @ W0 + b0), N=128
__global__ __launch_bounds__(256) void gemm_cls1(
    const _Float16* __restrict__ Xh, const int* __restrict__ src,
    const int* __restrict__ dst, const float* __restrict__ EA,
    const _Float16* __restrict__ Wt /*[128][544]*/, const float* __restrict__ bias,
    _Float16* __restrict__ out /*[Ep,128]*/) {
    const int lane  = threadIdx.x & 31;
    const int wave  = threadIdx.x >> 5;
    const int mrow  = blockIdx.x * 128 + wave * 16;
    const int ntile = blockIdx.y * 64;
    const int lm    = lane & 15;
    const int kg    = (lane >> 4) * 8;

    int e  = mrow + lm;
    int ec = (e < NE) ? e : (NE - 1);               // clamp pad rows
    const _Float16* xs = Xh + (size_t)src[ec] * HID;
    const _Float16* xd = Xh + (size_t)dst[ec] * HID;
    const float*    ea = EA + (size_t)ec * EF;
    const _Float16* brow = Wt + (size_t)(ntile + lm) * KC + kg;

    v8f acc[4];
#pragma unroll
    for (int t = 0; t < 4; ++t) acc[t] = {};

    v16h a  = loadA_concat(xs, xd, ea, kg);
    v16h b0 = loadfrag(brow);
    v16h b1 = loadfrag(brow + (size_t)16 * KC);
    v16h b2 = loadfrag(brow + (size_t)32 * KC);
    v16h b3 = loadfrag(brow + (size_t)48 * KC);

#pragma unroll
    for (int kk = 32; kk <= KC; kk += 32) {
        v16h an = a, bn0 = b0, bn1 = b1, bn2 = b2, bn3 = b3;
        if (kk < KC) {
            an  = loadA_concat(xs, xd, ea, kk + kg);
            bn0 = loadfrag(brow + kk);
            bn1 = loadfrag(brow + (size_t)16 * KC + kk);
            bn2 = loadfrag(brow + (size_t)32 * KC + kk);
            bn3 = loadfrag(brow + (size_t)48 * KC + kk);
        }
        acc[0] = __builtin_amdgcn_wmma_f32_16x16x32_f16(false, a, false, b0, (short)0, acc[0], false, false);
        acc[1] = __builtin_amdgcn_wmma_f32_16x16x32_f16(false, a, false, b1, (short)0, acc[1], false, false);
        acc[2] = __builtin_amdgcn_wmma_f32_16x16x32_f16(false, a, false, b2, (short)0, acc[2], false, false);
        acc[3] = __builtin_amdgcn_wmma_f32_16x16x32_f16(false, a, false, b3, (short)0, acc[3], false, false);
        a = an; b0 = bn0; b1 = bn1; b2 = bn2; b3 = bn3;
    }

    const int orow = mrow + (lane >> 4) * 8;
#pragma unroll
    for (int t = 0; t < 4; ++t) {
        int col = ntile + 16 * t + lm;
        float bv = bias[col];
#pragma unroll
        for (int v = 0; v < 8; ++v) {
            float x = fmaxf(acc[t][v] + bv, 0.0f);
            out[(size_t)(orow + v) * 128 + col] = (_Float16)x;
        }
    }
}

// ---------------------------------------------------------------- aggregation

// AGG[n,f] = h[n,f] * inv_deg[n]   (self-loop term), grid.x = NN, block = 256 = HID
__global__ void k_selfloop(const _Float16* __restrict__ Hh,
                           const float* __restrict__ idg, float* __restrict__ AGG) {
    int n = blockIdx.x, f = threadIdx.x;
    size_t i = (size_t)n * HID + f;
    AGG[i] = (float)Hh[i] * idg[n];
}

// AGG[dst,f] += h[src,f] * norm_e   grid.x = NE, block = 256 = HID
__global__ void k_scatter(const int* __restrict__ src, const int* __restrict__ dst,
                          const float* __restrict__ nrme,
                          const _Float16* __restrict__ Hh, float* __restrict__ AGG) {
    int e = blockIdx.x, f = threadIdx.x;
    int s = src[e], d = dst[e];
    float v = (float)Hh[(size_t)s * HID + f] * nrme[e];
    unsafeAtomicAdd(&AGG[(size_t)d * HID + f], v);
}

// x = relu(AGG) -> f16, pad rows zeroed.  grid.x = Mp, block = 256
__global__ void k_relu_cvt(const float* __restrict__ AGG, _Float16* __restrict__ Xh) {
    int row = blockIdx.x;
    size_t i = (size_t)row * HID + threadIdx.x;
    Xh[i] = (row < NN) ? (_Float16)fmaxf(AGG[i], 0.0f) : (_Float16)0.0f;
}

// ---------------------------------------------------------------- final layer (N=2)

__global__ void k_final(const _Float16* __restrict__ H2, const float* __restrict__ W2,
                        const float* __restrict__ b2, float* __restrict__ out) {
    int e = blockIdx.x * 256 + threadIdx.x;
    if (e >= NE) return;
    const _Float16* h = H2 + (size_t)e * 64;
    float a0 = b2[0], a1 = b2[1];
#pragma unroll 8
    for (int k = 0; k < 64; ++k) {
        float v = (float)h[k];
        a0 += v * W2[k * 2 + 0];
        a1 += v * W2[k * 2 + 1];
    }
    out[(size_t)e * 2 + 0] = a0;
    out[(size_t)e * 2 + 1] = a1;
}

// ---------------------------------------------------------------- host

extern "C" void kernel_launch(void* const* d_in, const int* in_sizes, int n_in,
                              void* d_out, int out_size, void* d_ws, size_t ws_size,
                              hipStream_t stream) {
    const float* nodef = (const float*)d_in[0];
    const int*   ei    = (const int*)d_in[1];
    const float* eattr = (const float*)d_in[2];
    const float* encW  = (const float*)d_in[3];
    const float* encB  = (const float*)d_in[4];
    const float* gW[3] = {(const float*)d_in[5], (const float*)d_in[7], (const float*)d_in[9]};
    const float* gB[3] = {(const float*)d_in[6], (const float*)d_in[8], (const float*)d_in[10]};
    const float* cW0 = (const float*)d_in[11]; const float* cB0 = (const float*)d_in[12];
    const float* cW1 = (const float*)d_in[13]; const float* cB1 = (const float*)d_in[14];
    const float* cW2 = (const float*)d_in[15]; const float* cB2 = (const float*)d_in[16];
    const int* src = ei;
    const int* dst = ei + NE;
    float* out = (float*)d_out;

    char* ws = (char*)d_ws;
    size_t off = 0;
    auto alloc = [&](size_t b) { size_t r = off; off = (off + b + 255) & ~(size_t)255; return r; };

    float*    deg  = (float*)(ws + alloc((size_t)NN * 4));
    float*    dinv = (float*)(ws + alloc((size_t)NN * 4));
    float*    idg  = (float*)(ws + alloc((size_t)NN * 4));
    float*    nrme = (float*)(ws + alloc((size_t)NE * 4));
    _Float16* wtE  = (_Float16*)(ws + alloc((size_t)HID * NF * 2));
    _Float16* wtG0 = (_Float16*)(ws + alloc((size_t)HID * HID * 2));
    _Float16* wtG1 = (_Float16*)(ws + alloc((size_t)HID * HID * 2));
    _Float16* wtG2 = (_Float16*)(ws + alloc((size_t)HID * HID * 2));
    _Float16* wtC0 = (_Float16*)(ws + alloc((size_t)128 * KC * 2));
    _Float16* wtC1 = (_Float16*)(ws + alloc((size_t)64 * 128 * 2));
    _Float16* NFh  = (_Float16*)(ws + alloc((size_t)Mp * NF * 2));  // 12.8 MB
    _Float16* Xh   = (_Float16*)(ws + alloc((size_t)Mp * HID * 2)); // 25.6 MB
    _Float16* Hh   = (_Float16*)(ws + alloc((size_t)Mp * HID * 2)); // 25.6 MB
    float*    AGG  = (float*)   (ws + alloc((size_t)Mp * HID * 4)); // 51.2 MB
    // aliases for the classifier stage (regions above are dead by then)
    _Float16* H1h = (_Float16*)Hh;   // needs Ep*128*2 = 76.8 MB <= Hh+AGG
    _Float16* H2h = (_Float16*)NFh;  // needs Ep*64*2  = 38.4 MB <= NFh+Xh
    _Float16* wtG[3] = {wtG0, wtG1, wtG2};

    // --- degree / norm prep
    k_fill_deg   <<<(NN + 255) / 256, 256, 0, stream>>>(deg);
    k_deg_scatter<<<(NE + 255) / 256, 256, 0, stream>>>(dst, deg);
    k_node_norm  <<<(NN + 255) / 256, 256, 0, stream>>>(deg, dinv, idg);
    k_edge_norm  <<<(NE + 255) / 256, 256, 0, stream>>>(src, dst, dinv, nrme);

    // --- weight transpose + f16 conversion
    k_wt<<<(HID * NF  + 255) / 256, 256, 0, stream>>>(encW, wtE, NF, HID, NF);
    for (int i = 0; i < 3; ++i)
        k_wt<<<(HID * HID + 255) / 256, 256, 0, stream>>>(gW[i], wtG[i], HID, HID, HID);
    k_wt<<<(128 * KC  + 255) / 256, 256, 0, stream>>>(cW0, wtC0, 2 * HID + EF, 128, KC);
    k_wt<<<(64  * 128 + 255) / 256, 256, 0, stream>>>(cW1, wtC1, 128, 64, 128);

    // --- node features -> f16
    {
        long total = (long)Mp * NF;
        k_cvt_rows<<<(int)((total + 255) / 256), 256, 0, stream>>>(nodef, NFh, NN, NF, total);
    }

    // --- encoder: Xh = relu(NF @ encW + b)
    gemm_f16<NF, true><<<dim3(Mp / 128, HID / 64), 256, 0, stream>>>(NFh, wtE, encB, Xh, HID);

    // --- 3 GCN layers
    for (int l = 0; l < 3; ++l) {
        gemm_f16<HID, false><<<dim3(Mp / 128, HID / 64), 256, 0, stream>>>(Xh, wtG[l], gB[l], Hh, HID);
        k_selfloop<<<NN, HID, 0, stream>>>(Hh, idg, AGG);
        k_scatter <<<NE, HID, 0, stream>>>(src, dst, nrme, Hh, AGG);
        k_relu_cvt<<<Mp, HID, 0, stream>>>(AGG, Xh);
    }

    // --- edge classifier
    gemm_cls1<<<dim3(Ep / 128, 128 / 64), 256, 0, stream>>>(Xh, src, dst, eattr, wtC0, cB0, H1h);
    gemm_f16<128, true><<<dim3(Ep / 128, 64 / 64), 256, 0, stream>>>(H1h, wtC1, cB1, H2h, 64);
    k_final<<<(NE + 255) / 256, 256, 0, stream>>>(H2h, cW2, cB2, out);
}
